// WaveletDownsample_63505386438696
// MI455X (gfx1250) — matrix operand order repaired and checked by reading, not compiled, not run on hardware
//
#include <hip/hip_runtime.h>
#include <hip/hip_bf16.h>

// ---------------------------------------------------------------------------
// WaveletDownsample fused kernel for gfx1250 (MI455X)
//   y = concat(Haar-DWT subbands)(x) * 0.5 ; out = conv3x3(y, w) + bias
// Implicit GEMM on v_wmma_f32_16x16x32_bf16 (wave32).
// ---------------------------------------------------------------------------

typedef __attribute__((ext_vector_type(8)))  float   v8f;
typedef __attribute__((ext_vector_type(8)))  __bf16  v8bf;
typedef __attribute__((ext_vector_type(16))) __bf16  v16bf;

#define NB    4      // batch
#define SRC_C 64     // channels of x
#define HIN   512
#define WIN   512
#define OC    64     // output channels
#define CIN   256    // y channels = 4 quads * 64
#define HO    256
#define WO    256

// ---------------------------------------------------------------------------
// Kernel A: repack weights fp32 (OC, CIN, 3, 3) OIHW  ->  bf16 workspace
// layout: [icb(8)][tap(9)][oc(64)][k_local(32)]
// K reordering: k_local = quad*8 + sb  where cin = quad*64 + (icb*8 + sb)
// ---------------------------------------------------------------------------
__global__ void wdw_repack_weights(const float* __restrict__ w,
                                   __bf16* __restrict__ wb) {
    int idx = blockIdx.x * 256 + threadIdx.x;         // 0 .. 147455 exact
    int oc  = idx / (CIN * 9);
    int rem = idx % (CIN * 9);
    int cin = rem / 9;
    int tap = rem % 9;
    int quad = cin >> 6;
    int src  = cin & 63;
    int icb  = src >> 3;
    int sb   = src & 7;
    int kl   = quad * 8 + sb;
    int dst  = ((icb * 9 + tap) * OC + oc) * 32 + kl;
    wb[dst] = (__bf16)w[idx];
}

// ---------------------------------------------------------------------------
// Kernel B: fused DWT + implicit GEMM conv.
// Grid: (WO/128, HO, NB), 256 threads = 8 wave32.
// Each block: output tile  M=64 (all oc) x N=128 (w segment) at row h.
// Wave (0..7): M-offset mo = 16*(wv>>1), N-offset nw = 64*(wv&1),
//              4 sub-tiles of 16 N each -> 4 v8f accumulators.
// ---------------------------------------------------------------------------
__global__ __launch_bounds__(256)
void wdw_dwt_conv_wmma(const float* __restrict__ x,
                       const __bf16* __restrict__ wb,
                       const float* __restrict__ bias,
                       float* __restrict__ out) {
    // 32B alignment: these are accessed as uint4 / v16bf vectors.
    __shared__ __align__(32) __bf16 Wlds[9 * OC * 32];    // 36,864 B [tap][oc][k]
    __shared__ __align__(32) __bf16 Ytile[3 * 130 * 32];  // 24,960 B [row][wl][k]

    const int tid  = threadIdx.x;
    const int lane = tid & 31;
    const int wv   = tid >> 5;
    const int half = lane >> 4;       // 0 | 1
    const int l15  = lane & 15;
    const int mo   = (wv >> 1) * 16;  // M offset of this wave
    const int nw   = (wv & 1) * 64;   // N offset of this wave

    const int w0 = blockIdx.x * 128;  // output w segment start
    const int h  = blockIdx.y;        // output row
    const int b  = blockIdx.z;        // batch

    v8f acc0 = {}, acc1 = {}, acc2 = {}, acc3 = {};

    for (int icb = 0; icb < 8; ++icb) {
        // ---- stage weight K-slice: contiguous 18432 bf16 = 2304 uint4 ----
        {
            const uint4* s = (const uint4*)(wb + (size_t)icb * (9 * OC * 32));
            uint4* d = (uint4*)Wlds;
            #pragma unroll
            for (int i = 0; i < 9; ++i) d[tid + i * 256] = s[tid + i * 256];
        }
        // Prefetch next weight K-slice into cache while we do DWT + WMMA.
        if (icb < 7) {
            __builtin_prefetch(wb + (size_t)(icb + 1) * (9 * OC * 32)
                                  + tid * 64, 0, 0);
        }
        // ---- fused Haar DWT -> LDS tile: 3 rows x 130 cols x 32 k -------
        // One x quad load yields 4 subband values (k = quad*8 + sb).
        for (int l = tid; l < 3 * 130 * 8; l += 256) {
            int wl  = l % 130;
            int t   = l / 130;           // 0..23 = row*8 + sb
            int sb  = t & 7;
            int row = t >> 3;
            int src = icb * 8 + sb;
            int yy  = h - 1 + row;       // y-space row (conv pad 1)
            int xx  = w0 - 1 + wl;       // y-space col
            float r0 = 0.f, r1 = 0.f, r2 = 0.f, r3 = 0.f;
            if ((unsigned)yy < 256u && (unsigned)xx < 256u) {
                const float* xp = x + (((size_t)(b * SRC_C + src) * HIN
                                        + 2 * yy) * WIN + 2 * xx);
                float a  = xp[0];
                float bb = xp[1];
                float c  = xp[WIN];
                float d  = xp[WIN + 1];
                r0 = ( a + bb + c + d) * 0.25f;   // LL * 0.5 * 0.5
                r1 = (-a - bb + c + d) * 0.25f;   // LH
                r2 = (-a + bb - c + d) * 0.25f;   // HL
                r3 = ( a - bb - c + d) * 0.25f;   // HH
            }
            __bf16* yp = &Ytile[(row * 130 + wl) * 32 + sb];
            yp[0]  = (__bf16)r0;
            yp[8]  = (__bf16)r1;
            yp[16] = (__bf16)r2;
            yp[24] = (__bf16)r3;
        }
        __syncthreads();

        // ---- 9 taps x 4 N-subtiles of WMMA --------------------------------
        #pragma unroll
        for (int tap = 0; tap < 9; ++tap) {
            const int kh = tap / 3, kw = tap % 3;
            // A fragment (16x32 bf16): lane holds M=l15,
            //   K = 16*half + 0..15
            const __bf16* ap = &Wlds[(tap * OC + mo + l15) * 32 + 16 * half];
            union { v16bf v; v8bf h2[2]; } af;
            af.h2[0] = *(const v8bf*)ap;
            af.h2[1] = *(const v8bf*)(ap + 8);
            // B fragments (32x16 bf16): lane holds N=l15, K = 16*half + 0..15
            const int rbase = (kh * 130 + nw + l15 + kw) * 32 + 16 * half;
            v16bf b0 = *(const v16bf*)&Ytile[rbase];
            v16bf b1 = *(const v16bf*)&Ytile[rbase + 512];   // +16 w
            v16bf b2 = *(const v16bf*)&Ytile[rbase + 1024];  // +32 w
            v16bf b3 = *(const v16bf*)&Ytile[rbase + 1536];  // +48 w
            acc0 = __builtin_amdgcn_wmma_f32_16x16x32_bf16(
                       false, af.v, false, b0, (short)0, acc0, false, false);
            acc1 = __builtin_amdgcn_wmma_f32_16x16x32_bf16(
                       false, af.v, false, b1, (short)0, acc1, false, false);
            acc2 = __builtin_amdgcn_wmma_f32_16x16x32_bf16(
                       false, af.v, false, b2, (short)0, acc2, false, false);
            acc3 = __builtin_amdgcn_wmma_f32_16x16x32_bf16(
                       false, af.v, false, b3, (short)0, acc3, false, false);
        }
        __syncthreads();
    }

    // ---- epilogue: C/D layout lane=N (l15), VGPR r -> M = r + 8*half ------
    #pragma unroll
    for (int s = 0; s < 4; ++s) {
        v8f a = (s == 0) ? acc0 : (s == 1) ? acc1 : (s == 2) ? acc2 : acc3;
        int wcol = w0 + nw + s * 16 + l15;
        #pragma unroll
        for (int r = 0; r < 8; ++r) {
            int oc = mo + r + 8 * half;
            out[(((size_t)b * OC + oc) * HO + h) * WO + wcol] = a[r] + bias[oc];
        }
    }
}

// ---------------------------------------------------------------------------
extern "C" void kernel_launch(void* const* d_in, const int* in_sizes, int n_in,
                              void* d_out, int out_size, void* d_ws, size_t ws_size,
                              hipStream_t stream) {
    const float* x    = (const float*)d_in[0];   // (4,64,512,512) f32
    const float* w    = (const float*)d_in[1];   // (64,256,3,3)  f32
    const float* bias = (const float*)d_in[2];   // (64,)         f32
    float* out        = (float*)d_out;           // (4,64,256,256) f32
    __bf16* wb        = (__bf16*)d_ws;           // 294,912 B of scratch

    // 147,456 weight elements, one thread each
    wdw_repack_weights<<<576, 256, 0, stream>>>(w, wb);

    dim3 grid(WO / 128, HO, NB);                 // (2, 256, 4)
    wdw_dwt_conv_wmma<<<grid, 256, 0, stream>>>(x, wb, bias, out);
}
